// ReverseMoE_3453153706590
// MI455X (gfx1250) — compile-verified, matrix operand order checked by reading, not run on hardware
//
#include <hip/hip_runtime.h>
#include <hip/hip_bf16.h>
#include <stdint.h>

typedef __attribute__((ext_vector_type(16))) __bf16 v16bf;
typedef __attribute__((ext_vector_type(8)))  float  v8f;

__device__ __forceinline__ unsigned short f2bf(float f) {
  unsigned u = __builtin_bit_cast(unsigned, f);
  u += 0x7FFFu + ((u >> 16) & 1u);          // round-to-nearest-even
  return (unsigned short)(u >> 16);
}

__device__ __forceinline__ float gelu_tanh(float x) {
  float x3 = x * x * x;
  float t  = tanhf(0.7978845608028654f * (x + 0.044715f * x3));
  return 0.5f * x * (1.0f + t);
}

// ---------------------------------------------------------------- gating
__global__ __launch_bounds__(128)
void gate_softmax_kernel(const float* __restrict__ xf, const float* __restrict__ Ws,
                         const float* __restrict__ bs, float* __restrict__ probT,
                         int N, int D, int E) {
  int tok  = blockIdx.x * 4 + (threadIdx.x >> 5);
  int lane = threadIdx.x & 31;
  const float* xr = xf + (size_t)tok * D;
  float acc[8] = {0.f,0.f,0.f,0.f,0.f,0.f,0.f,0.f};
  for (int d = lane; d < D; d += 32) {
    float xv = xr[d];
    const float4 w0 = *(const float4*)(Ws + d * 8);
    const float4 w1 = *(const float4*)(Ws + d * 8 + 4);
    acc[0] += xv * w0.x; acc[1] += xv * w0.y; acc[2] += xv * w0.z; acc[3] += xv * w0.w;
    acc[4] += xv * w1.x; acc[5] += xv * w1.y; acc[6] += xv * w1.z; acc[7] += xv * w1.w;
  }
#pragma unroll
  for (int off = 16; off > 0; off >>= 1) {
#pragma unroll
    for (int e = 0; e < 8; ++e) acc[e] += __shfl_xor(acc[e], off, 32);
  }
  float mx = -1e30f;
#pragma unroll
  for (int e = 0; e < 8; ++e) { acc[e] += bs[e]; mx = fmaxf(mx, acc[e]); }
  float s = 0.f;
#pragma unroll
  for (int e = 0; e < 8; ++e) { acc[e] = expf(acc[e] - mx); s += acc[e]; }
  float inv = 1.0f / s;
  if (lane < E) probT[(size_t)lane * N + tok] = acc[lane] * inv;
}

// ---------------------------------------------------------------- top-k
__global__ __launch_bounds__(512)
void topk_kernel(const float* __restrict__ probT, float* __restrict__ vals,
                 int* __restrict__ routes, int N, int cap) {
  __shared__ unsigned long long keys[4096];
  const int e = blockIdx.x;
  const float* row = probT + (size_t)e * N;
  for (int i = threadIdx.x; i < 4096; i += 512) {
    unsigned long long kk = 0ull;
    if (i < N) {
      unsigned u = __builtin_bit_cast(unsigned, row[i]);
      u = (u & 0x80000000u) ? ~u : (u | 0x80000000u);
      kk = ((unsigned long long)u << 32) | (unsigned)(~i);
    }
    keys[i] = kk;
  }
  __syncthreads();
  for (int k = 2; k <= 4096; k <<= 1) {
    for (int j = k >> 1; j > 0; j >>= 1) {
      for (int i = threadIdx.x; i < 4096; i += 512) {
        int p = i ^ j;
        if (p > i) {
          unsigned long long a = keys[i], b = keys[p];
          bool up = ((i & k) == 0);          // descending overall
          if (up ? (a < b) : (a > b)) { keys[i] = b; keys[p] = a; }
        }
      }
      __syncthreads();
    }
  }
  for (int i = threadIdx.x; i < cap; i += 512) {
    unsigned long long kk = keys[i];
    unsigned u  = (unsigned)(kk >> 32);
    unsigned fb = (u & 0x80000000u) ? (u ^ 0x80000000u) : ~u;
    vals[(size_t)e * cap + i]   = __builtin_bit_cast(float, fb);
    routes[(size_t)e * cap + i] = (int)(~(unsigned)(kk & 0xffffffffu));
  }
}

// ---------------------------------------------------------------- gather
__global__ __launch_bounds__(256)
void gather_bf16_kernel(const float* __restrict__ xf, const int* __restrict__ routes,
                        unsigned short* __restrict__ xin, int D) {
  int row = blockIdx.x;
  int tok = routes[row];
  const float* src = xf + (size_t)tok * D;
  unsigned short* dst = xin + (size_t)row * D;
  for (int c = threadIdx.x * 4; c < D; c += blockDim.x * 4) {
    float4 v = *(const float4*)(src + c);
    uint2 pk;
    pk.x = (unsigned)f2bf(v.x) | ((unsigned)f2bf(v.y) << 16);
    pk.y = (unsigned)f2bf(v.z) | ((unsigned)f2bf(v.w) << 16);
    *(uint2*)(dst + c) = pk;
  }
}

// ---------------------------------------------------------------- grouped GEMM
// 128x128 block tile, K-step 32, 8 waves of 32x64 (2x4 WMMA tiles each).
// Double-buffered LDS, one barrier per K-step; next tile's global loads are
// issued before the current tile's WMMAs so loadcnt waits land after compute.
// LDS fragment swizzle (16-bit A 16x32 VGPR layout):
//   lane = r + 2*(k&8) ; j = (k&7) | ((k&16)>>1)
template<int EPI>
__global__ __launch_bounds__(256)
void moe_gemm(const unsigned short* __restrict__ A,   // [E, M, K] bf16 row-major
              const float* __restrict__ B,            // [E, K, N] f32 row-major
              const float* __restrict__ bias,         // [E, N]
              const float* __restrict__ vals,         // [E, M]   (EPI=1)
              const int* __restrict__ routes,         // [E, M]   (EPI=1)
              unsigned short* __restrict__ Hout,      // [E, M, N] bf16 (EPI=0)
              float* __restrict__ Out,                // [Ntok, N] f32  (EPI=1)
              int M, int N, int K) {
  __shared__ __align__(32) unsigned short lA[2 * 8 * 32 * 16];   // 16KB (2 bufs)
  __shared__ __align__(32) unsigned short lB[2 * 8 * 32 * 16];   // 16KB (2 bufs)

  const int e  = blockIdx.z;
  const int mb = blockIdx.y * 128;
  const int nb = blockIdx.x * 128;
  const unsigned short* Ae = A + (size_t)e * M * K;
  const float* Be = B + (size_t)e * K * N;
  const float* be = bias + (size_t)e * N;

  const int tid  = threadIdx.x;
  const int lane = tid & 31;
  const int wave = tid >> 5;
  const int wm = wave & 3;      // 4 waves along M
  const int wn = wave >> 2;     // 2 waves along N

  v8f acc[2][4];
#pragma unroll
  for (int i = 0; i < 2; ++i)
#pragma unroll
    for (int j = 0; j < 4; ++j) acc[i][j] = v8f{0.f,0.f,0.f,0.f,0.f,0.f,0.f,0.f};

  // A staging mapping: thread -> (row r, half of K32)
  const int ar = tid >> 1, ah = tid & 1;
  const int amt = ar >> 4, arl = ar & 15;
  // B staging mapping: thread -> (nt, lane-in-frag)
  const int bnt = tid >> 5, bl = tid & 31, bnl = bl & 15, bkhi = (bl >> 4) * 8;

  uint4 sa0, sa1;     // staged A (already bf16)
  float sb[16];       // staged B (f32, converted at LDS-store time)

  auto stage_load = [&](int k0) {
    const uint4* g = (const uint4*)(Ae + (size_t)(mb + ar) * K + k0 + ah * 16);
    sa0 = g[0];       // k = ah*16 + 0..7   -> lane arl,    j = ah*8..
    sa1 = g[1];       // k = ah*16 + 8..15  -> lane arl+16, j = ah*8..
    const float* gB = Be + (size_t)(k0 + bkhi) * N + nb + bnt * 16 + bnl;
#pragma unroll
    for (int j = 0; j < 8; ++j) sb[j]     = gB[(size_t)j * N];         // K = khi+j
#pragma unroll
    for (int j = 0; j < 8; ++j) sb[8 + j] = gB[(size_t)(16 + j) * N];  // K = 16+khi+j
  };

  auto stage_store = [&](int buf) {
    unsigned short* A_ = lA + buf * 4096;
    unsigned short* B_ = lB + buf * 4096;
    *(uint4*)&A_[((amt * 32 + arl)      * 16) + ah * 8] = sa0;
    *(uint4*)&A_[((amt * 32 + arl + 16) * 16) + ah * 8] = sa1;
    unsigned short tmp[16];
#pragma unroll
    for (int j = 0; j < 16; ++j) tmp[j] = f2bf(sb[j]);
    *(uint4*)&B_[(bnt * 32 + bl) * 16]     = *(uint4*)&tmp[0];
    *(uint4*)&B_[(bnt * 32 + bl) * 16 + 8] = *(uint4*)&tmp[8];
  };

  const int niter = K / 32;
  stage_load(0);
  stage_store(0);
  __syncthreads();

  for (int it = 0; it < niter; ++it) {
    const int cur = it & 1;
    const bool more = (it + 1 < niter);
    if (more) stage_load((it + 1) * 32);   // global loads first

    const unsigned short* A_ = lA + cur * 4096;
    const unsigned short* B_ = lB + cur * 4096;
    v16bf af[2], bfv[4];
#pragma unroll
    for (int i = 0; i < 2; ++i)
      af[i] = *(const v16bf*)&A_[((wm * 2 + i) * 32 + lane) * 16];
#pragma unroll
    for (int j = 0; j < 4; ++j)
      bfv[j] = *(const v16bf*)&B_[((wn * 4 + j) * 32 + lane) * 16];
#pragma unroll
    for (int i = 0; i < 2; ++i)
#pragma unroll
      for (int j = 0; j < 4; ++j)
        acc[i][j] = __builtin_amdgcn_wmma_f32_16x16x32_bf16(
            false, af[i], false, bfv[j], (short)0, acc[i][j], false, false);

    if (more) stage_store(1 - cur);        // write the buffer nobody reads
    __syncthreads();                       // protects next reads + cur overwrite
  }

  // ---- epilogue: VGPR r -> row = tile*16 + r + (lane>=16 ? 8:0),
  //                           col = tile*16 + (lane&15)
  const int rlo = (lane >> 4) << 3;
  const int cl  = lane & 15;
#pragma unroll
  for (int i = 0; i < 2; ++i) {
    const int mtile = wm * 2 + i;
    float  srow[8];
    int    trow[8];
    if (EPI == 1) {
#pragma unroll
      for (int r = 0; r < 8; ++r) {
        const int row = mb + mtile * 16 + rlo + r;
        srow[r] = vals[(size_t)e * M + row];
        trow[r] = routes[(size_t)e * M + row];
      }
    }
#pragma unroll
    for (int j = 0; j < 4; ++j) {
      const int ntile = wn * 4 + j;
      const int col = nb + ntile * 16 + cl;
      const float bv = be[col];
#pragma unroll
      for (int r = 0; r < 8; ++r) {
        const int row = mb + mtile * 16 + rlo + r;
        float v = acc[i][j][r] + bv;
        if (EPI == 0) {
          v = gelu_tanh(v);
          Hout[(size_t)e * M * N + (size_t)row * N + col] = f2bf(v);
        } else {
          __hip_atomic_fetch_add(&Out[(size_t)trow[r] * N + col], v * srow[r],
                                 __ATOMIC_RELAXED, __HIP_MEMORY_SCOPE_AGENT);
        }
      }
    }
  }
}

// ---------------------------------------------------------------- launch
extern "C" void kernel_launch(void* const* d_in, const int* in_sizes, int n_in,
                              void* d_out, int out_size, void* d_ws, size_t ws_size,
                              hipStream_t stream) {
  (void)n_in; (void)ws_size;
  const float* x  = (const float*)d_in[0];
  const float* Ws = (const float*)d_in[1];
  const float* bs = (const float*)d_in[2];
  const float* W1 = (const float*)d_in[3];
  const float* b1 = (const float*)d_in[4];
  const float* W2 = (const float*)d_in[5];
  const float* b2 = (const float*)d_in[6];
  float* out = (float*)d_out;

  const int E   = in_sizes[2];          // 8
  const int D   = in_sizes[1] / E;      // 1024
  const int F   = in_sizes[4] / E;      // 4096
  const int N   = in_sizes[0] / D;      // 4096 tokens
  const int cap = N / E;                // 512

  char* ws = (char*)d_ws;
  float* probT = (float*)ws;                                   // 128 KB
  float* vals  = (float*)(ws + (256 << 10));                   //  16 KB
  int*   routes = (int*)(ws + (320 << 10));                    //  16 KB
  unsigned short* xin = (unsigned short*)(ws + (1ull  << 20)); //   8 MB
  unsigned short* h   = (unsigned short*)(ws + (16ull << 20)); //  32 MB

  hipMemsetAsync(d_out, 0, (size_t)out_size * sizeof(float), stream);

  gate_softmax_kernel<<<N / 4, 128, 0, stream>>>(x, Ws, bs, probT, N, D, E);
  topk_kernel<<<E, 512, 0, stream>>>(probT, vals, routes, N, cap);
  gather_bf16_kernel<<<E * cap, 256, 0, stream>>>(x, routes, xin, D);

  dim3 g1(F / 128, cap / 128, E);       // (32, 4, 8)
  moe_gemm<0><<<g1, 256, 0, stream>>>(xin, W1, b1, nullptr, nullptr, h, nullptr,
                                      cap, F, D);
  dim3 g2(D / 128, cap / 128, E);       // (8, 4, 8)
  moe_gemm<1><<<g2, 256, 0, stream>>>(h, W2, b2, vals, routes, nullptr, out,
                                      cap, D, F);
}
